// SelfAttention_39831526703514
// MI455X (gfx1250) — matrix operand order repaired and checked by reading, not compiled, not run on hardware
//
#include <hip/hip_runtime.h>
#include <stdint.h>

#define B_  4
#define S_  2048
#define D_  1024
#define H_  8
#define DK_ 128
#define HDK_ (H_ * DK_)   // 1024

typedef __attribute__((ext_vector_type(16))) __bf16 v16bf;
typedef __attribute__((ext_vector_type(8)))  float  v8f;

union FragB { v16bf v; uint32_t u[8]; };

__device__ __forceinline__ uint16_t f2bf(float f) {
    uint32_t x = __float_as_uint(f);
    x += 0x7FFFu + ((x >> 16) & 1u);          // round-to-nearest-even
    return (uint16_t)(x >> 16);
}

__device__ __forceinline__ v8f wmma_bf16(const FragB& a, const FragB& b, v8f c) {
    return __builtin_amdgcn_wmma_f32_16x16x32_bf16(
        /*neg_a=*/false, a.v, /*neg_b=*/false, b.v,
        /*c_mod=*/(short)0, c, /*reuse_a=*/false, /*reuse_b=*/false);
}

// A-fragment (16x32 bf16) per-lane K offsets: lanes 0-15 hold K {0..15},
// lanes 16-31 hold K {16..31}, two per VGPR.
__device__ __forceinline__ void a_offsets(int hi, int ak[8]) {
#pragma unroll
    for (int v = 0; v < 4; ++v) {
        ak[v]     = hi * 16 + 2 * v;
        ak[v + 4] = hi * 16 + 8 + 2 * v;
    }
}

// ---------------- conversion kernels ----------------

__global__ void cvt_f32_bf16(const float* __restrict__ src,
                             uint16_t* __restrict__ dst, int n) {
    int i = blockIdx.x * blockDim.x + threadIdx.x;
    if (i < n) dst[i] = f2bf(src[i]);
}

// src: [H][D][DK]  ->  dst: [H][DK][D]  (bf16)
__global__ void cvt_w_transpose(const float* __restrict__ src,
                                uint16_t* __restrict__ dst) {
    int i = blockIdx.x * blockDim.x + threadIdx.x;
    if (i >= H_ * DK_ * D_) return;
    int d = i % D_;
    int t = i / D_;
    int n = t % DK_;
    int h = t / DK_;
    dst[i] = f2bf(src[(h * D_ + d) * DK_ + n]);
}

// ---------------- QKV projection ----------------
// One wave computes a 16(s) x 128(dk) tile of Q, K or V for one (b,h).
// K,Q stored [b][h][s][dk]; V stored transposed [b][h][dk][s].
__global__ void __launch_bounds__(256) qkv_proj(
    const uint16_t* __restrict__ xb,
    const uint16_t* __restrict__ Wqt,
    const uint16_t* __restrict__ Wkt,
    const uint16_t* __restrict__ Wvt,
    const float* __restrict__ bias,
    uint16_t* __restrict__ Qb,
    uint16_t* __restrict__ Kb,
    uint16_t* __restrict__ Vtb) {
    const int S16 = S_ / 16;
    int wave = (int)((blockIdx.x * blockDim.x + threadIdx.x) >> 5);
    int lane = threadIdx.x & 31;
    int row  = lane & 15, hi = lane >> 4;

    int st = wave % S16;  int r = wave / S16;
    int h  = r % H_;      r /= H_;
    int b  = r % B_;      int which = r / B_;
    if (which >= 3) return;

    const uint16_t* Wt = (which == 0) ? Wqt : (which == 1) ? Wkt : Wvt;
    int s0 = st * 16;
    const uint16_t* Arow = xb + (size_t)(b * S_ + s0 + row) * D_;
    const uint16_t* Wh   = Wt + (size_t)h * DK_ * D_;

    int ak[8]; a_offsets(hi, ak);

    v8f acc[8];
#pragma unroll
    for (int nt = 0; nt < 8; ++nt) acc[nt] = {};

#pragma unroll 1
    for (int kk = 0; kk < D_; kk += 32) {
        if (kk + 128 < D_) __builtin_prefetch(Arow + kk + 128, 0, 1);
        FragB a;
#pragma unroll
        for (int v = 0; v < 8; ++v)
            a.u[v] = *(const uint32_t*)(Arow + kk + ak[v]);
#pragma unroll
        for (int nt = 0; nt < 8; ++nt) {
            const uint16_t* Wrow = Wh + (size_t)(nt * 16 + row) * D_ + kk + hi * 16;
            FragB bf;
#pragma unroll
            for (int v = 0; v < 8; ++v)
                bf.u[v] = *(const uint32_t*)(Wrow + 2 * v);
            acc[nt] = wmma_bf16(a, bf, acc[nt]);
        }
    }

    float bb = bias[0];
    size_t bh = (size_t)(b * H_ + h);
#pragma unroll
    for (int nt = 0; nt < 8; ++nt) {
#pragma unroll
        for (int v = 0; v < 8; ++v) {
            int s   = s0 + hi * 8 + v;          // C layout: M = v (+8 for hi half)
            int col = nt * 16 + row;            // N = lane%16
            uint16_t val = f2bf(acc[nt][v] + bb);
            if (which == 2)
                Vtb[(bh * DK_ + col) * S_ + s] = val;
            else if (which == 0)
                Qb[(bh * S_ + s) * DK_ + col] = val;
            else
                Kb[(bh * S_ + s) * DK_ + col] = val;
        }
    }
}

// ---------------- flash attention ----------------
// 1 wave per (b,h,16-row query tile); 4 independent waves per block.
__global__ void __launch_bounds__(128) flash_attn(
    const uint16_t* __restrict__ Qb,
    const uint16_t* __restrict__ Kb,
    const uint16_t* __restrict__ Vtb,
    const int* __restrict__ mask,
    uint16_t* __restrict__ Hb) {
    __shared__ __align__(16) uint16_t pshared[4 * 16 * 32];

    const int S16 = S_ / 16;
    int warp = threadIdx.x >> 5;
    int lane = threadIdx.x & 31;
    int row  = lane & 15, hi = lane >> 4;

    int job = blockIdx.x * 4 + warp;
    int st  = job % S16;  int r = job / S16;
    int h   = r % H_;     int b = r / H_;
    int s0  = st * 16;
    size_t bh = (size_t)(b * H_ + h);

    int ak[8]; a_offsets(hi, ak);

    // Q fragments (16 x 128 = 4 chunks of 16x32), reused for all key tiles.
    FragB q[4];
    const uint16_t* Qrow = Qb + (bh * S_ + s0 + row) * DK_;
#pragma unroll
    for (int i = 0; i < 4; ++i)
#pragma unroll
        for (int v = 0; v < 8; ++v)
            q[i].u[v] = *(const uint32_t*)(Qrow + 32 * i + ak[v]);

    v8f acc[8];
    float m_run[8], l_run[8];
#pragma unroll
    for (int nt = 0; nt < 8; ++nt) acc[nt] = {};
#pragma unroll
    for (int v = 0; v < 8; ++v) { m_run[v] = -3.0e38f; l_run[v] = 0.0f; }

    uint16_t* pb = pshared + warp * (16 * 32);
    const float scale = 0.088388347648318447f;   // 1/sqrt(128)
    const int* mrow = mask + b * S_;

#pragma unroll 1
    for (int t0 = 0; t0 < S_; t0 += 32) {
        // ---- scores: two 16x16 tiles over keys [t0,t0+16) and [t0+16,t0+32)
        v8f c0 = {}, c1 = {};
        const uint16_t* K0 = Kb + (bh * S_ + t0 + row) * DK_ + hi * 16;
        const uint16_t* K1 = K0 + 16 * DK_;
#pragma unroll
        for (int i = 0; i < 4; ++i) {
            FragB bk;
#pragma unroll
            for (int v = 0; v < 8; ++v)
                bk.u[v] = *(const uint32_t*)(K0 + 32 * i + 2 * v);
            c0 = wmma_bf16(q[i], bk, c0);
#pragma unroll
            for (int v = 0; v < 8; ++v)
                bk.u[v] = *(const uint32_t*)(K1 + 32 * i + 2 * v);
            c1 = wmma_bf16(q[i], bk, c1);
        }

        float madd0 = mrow[t0 + row]      ? 0.0f : -1.0e30f;
        float madd1 = mrow[t0 + 16 + row] ? 0.0f : -1.0e30f;

        // ---- online softmax (rows live in 16-lane halves; xor 1/2/4/8 stays in-half)
        float p0[8], p1[8], alpha[8];
#pragma unroll
        for (int v = 0; v < 8; ++v) {
            float x0 = c0[v] * scale + madd0;
            float x1 = c1[v] * scale + madd1;
            float tm = fmaxf(x0, x1);
            tm = fmaxf(tm, __shfl_xor(tm, 1));
            tm = fmaxf(tm, __shfl_xor(tm, 2));
            tm = fmaxf(tm, __shfl_xor(tm, 4));
            tm = fmaxf(tm, __shfl_xor(tm, 8));
            float mn = fmaxf(m_run[v], tm);
            float al = __expf(m_run[v] - mn);
            float e0 = __expf(x0 - mn);
            float e1 = __expf(x1 - mn);
            float rs = e0 + e1;
            rs += __shfl_xor(rs, 1);
            rs += __shfl_xor(rs, 2);
            rs += __shfl_xor(rs, 4);
            rs += __shfl_xor(rs, 8);
            l_run[v] = al * l_run[v] + rs;
            m_run[v] = mn;
            alpha[v] = al;
            p0[v] = e0;  p1[v] = e1;
        }
#pragma unroll
        for (int nt = 0; nt < 8; ++nt)
#pragma unroll
            for (int v = 0; v < 8; ++v) acc[nt][v] *= alpha[v];

        // ---- P -> LDS (C layout) -> A-fragment reload
        asm volatile("" ::: "memory");
#pragma unroll
        for (int v = 0; v < 8; ++v) {
            int m = hi * 8 + v;
            pb[m * 32 + row]      = f2bf(p0[v]);
            pb[m * 32 + 16 + row] = f2bf(p1[v]);
        }
        asm volatile("s_wait_dscnt 0x0" ::: "memory");
        FragB pf;
#pragma unroll
        for (int v = 0; v < 8; ++v)
            pf.u[v] = *(const uint32_t*)(pb + row * 32 + ak[v]);

        // ---- acc += P (16x32) x V (32x128); V is transposed so loads are dwords
#pragma unroll
        for (int nt = 0; nt < 8; ++nt) {
            const uint16_t* Vr = Vtb + (bh * DK_ + nt * 16 + row) * S_ + t0 + hi * 16;
            FragB vb;
#pragma unroll
            for (int v = 0; v < 8; ++v)
                vb.u[v] = *(const uint32_t*)(Vr + 2 * v);
            acc[nt] = wmma_bf16(pf, vb, acc[nt]);
        }
    }

    // ---- epilogue: normalize, reverse head order, store bf16
    int hrev = H_ - 1 - h;
#pragma unroll
    for (int v = 0; v < 8; ++v) {
        float inv = 1.0f / l_run[v];
        int s = s0 + hi * 8 + v;
        uint16_t* Ho = Hb + (size_t)(b * S_ + s) * HDK_ + hrev * DK_;
#pragma unroll
        for (int nt = 0; nt < 8; ++nt)
            Ho[nt * 16 + row] = f2bf(acc[nt][v] * inv);
    }
}

// ---------------- output projection: out = MH @ Wo^T + b ----------------
// 1 wave = one 16-row x 64-col tile (4 accumulators, A reused).
__global__ void __launch_bounds__(256) out_proj(
    const uint16_t* __restrict__ Hb,
    const uint16_t* __restrict__ Wob,
    const float* __restrict__ bias,
    float* __restrict__ out) {
    int wave = (int)((blockIdx.x * blockDim.x + threadIdx.x) >> 5);
    int lane = threadIdx.x & 31;
    int row  = lane & 15, hi = lane >> 4;

    int mt = wave >> 4;         // 512 row tiles of (B*S)
    int ng = wave & 15;         // 16 col groups of 64
    int n0 = ng * 64;

    int ak[8]; a_offsets(hi, ak);
    const uint16_t* Arow = Hb + (size_t)(mt * 16 + row) * HDK_;

    v8f acc[4];
#pragma unroll
    for (int nt = 0; nt < 4; ++nt) acc[nt] = {};

#pragma unroll 1
    for (int kk = 0; kk < HDK_; kk += 32) {
        if (kk + 128 < HDK_) __builtin_prefetch(Arow + kk + 128, 0, 1);
        FragB a;
#pragma unroll
        for (int v = 0; v < 8; ++v)
            a.u[v] = *(const uint32_t*)(Arow + kk + ak[v]);
#pragma unroll
        for (int nt = 0; nt < 4; ++nt) {
            // B element (k=j, n=d) = Wo[d][j]; rows of Wo are contiguous in j.
            const uint16_t* Wr = Wob + (size_t)(n0 + nt * 16 + row) * HDK_ + kk + hi * 16;
            FragB bf;
#pragma unroll
            for (int v = 0; v < 8; ++v)
                bf.u[v] = *(const uint32_t*)(Wr + 2 * v);
            acc[nt] = wmma_bf16(a, bf, acc[nt]);
        }
    }

    float bb = bias[0];
#pragma unroll
    for (int nt = 0; nt < 4; ++nt)
#pragma unroll
        for (int v = 0; v < 8; ++v)
            out[(size_t)(mt * 16 + hi * 8 + v) * D_ + n0 + nt * 16 + row] =
                acc[nt][v] + bb;
}

// ---------------- launcher ----------------
extern "C" void kernel_launch(void* const* d_in, const int* in_sizes, int n_in,
                              void* d_out, int out_size, void* d_ws, size_t ws_size,
                              hipStream_t stream) {
    const float* x    = (const float*)d_in[0];
    const int*   mask = (const int*)  d_in[1];
    const float* W_q  = (const float*)d_in[2];
    const float* W_k  = (const float*)d_in[3];
    const float* W_v  = (const float*)d_in[4];
    const float* W_o  = (const float*)d_in[5];
    const float* bias = (const float*)d_in[6];
    float* out = (float*)d_out;

    const size_t nX = (size_t)B_ * S_ * D_;        // 8,388,608
    const size_t nW = (size_t)H_ * D_ * DK_;       // 1,048,576
    const size_t nQ = (size_t)B_ * H_ * S_ * DK_;  // 8,388,608

    uint16_t* w   = (uint16_t*)d_ws;
    uint16_t* xb  = w;  w += nX;
    uint16_t* Wqt = w;  w += nW;
    uint16_t* Wkt = w;  w += nW;
    uint16_t* Wvt = w;  w += nW;
    uint16_t* Wob = w;  w += nW;
    uint16_t* Qb  = w;  w += nQ;
    uint16_t* Kb  = w;  w += nQ;
    uint16_t* Vtb = w;  w += nQ;
    uint16_t* Hb  = w;  w += nX;   // B*S*HDK == nX

    cvt_f32_bf16   <<<(int)((nX + 255) / 256), 256, 0, stream>>>(x,  xb,  (int)nX);
    cvt_w_transpose<<<(int)((nW + 255) / 256), 256, 0, stream>>>(W_q, Wqt);
    cvt_w_transpose<<<(int)((nW + 255) / 256), 256, 0, stream>>>(W_k, Wkt);
    cvt_w_transpose<<<(int)((nW + 255) / 256), 256, 0, stream>>>(W_v, Wvt);
    cvt_f32_bf16   <<<(int)((nW + 255) / 256), 256, 0, stream>>>(W_o, Wob, (int)nW);

    // 3 * B * H * (S/16) waves = 12288 waves -> 1536 blocks of 256
    qkv_proj <<<1536, 256, 0, stream>>>(xb, Wqt, Wkt, Wvt, bias, Qb, Kb, Vtb);
    // B * H * (S/16) = 4096 waves, 4 per block -> 1024 blocks of 128
    flash_attn<<<1024, 128, 0, stream>>>(Qb, Kb, Vtb, mask, Hb);
    // (B*S/16) * 16 = 8192 waves -> 1024 blocks of 256
    out_proj <<<1024, 256, 0, stream>>>(Hb, Wob, bias, out);
}
